// ToLogMelSpec_76940044140612
// MI455X (gfx1250) — matrix operand order repaired and checked by reading, not compiled, not run on hardware
//
#include <hip/hip_runtime.h>
#include <math.h>

typedef __attribute__((ext_vector_type(16))) _Float16 v16h;
typedef __attribute__((ext_vector_type(8)))  float    v8f;
typedef __attribute__((ext_vector_type(4)))  unsigned int uint32x4;
typedef __attribute__((ext_vector_type(8)))  int      int32x8;
typedef __attribute__((ext_vector_type(4)))  int      int32x4;

#define N_FFT    400
#define HOP      160
#define N_FREQ   201
#define N_MELS   80
#define N_FRAMES 1001
#define T_LEN    160000
#define EPS      1.1920929e-07f

#define FPB      64                      // frames per workgroup
#define NWAVE    4                       // 128 threads
#define KT       13                      // K tiles of 32  -> 416 >= 400
#define NT       13                      // freq tiles of 16 -> 208 >= 201
#define MKT      7                       // mel K tiles of 32 -> 224 >= 208
#define MNT      5                       // mel N tiles of 16 -> 80
#define AUD_LDS  ((FPB - 1) * HOP + N_FFT)   // 10480 floats
#define PFREQ    224                     // padded freq dim in LDS

#define TILE_HALFS  (KT * 512)                   // 6656 halves = 13312 B per (tile)
#define DFT_FRAG_HALFS (2 * NT * TILE_HALFS)     // 173056 halves (~338 KB)
#define MEL_FRAG_HALFS (MNT * MKT * 32 * 16)     // 17920 halves  (~35 KB)

// ---------------------------------------------------------------------------
// TDM: 1D tile load, global -> LDS.  n8 = length in 8-byte units.
// D# packing per CDNA5 ISA ch.8 (group0: count/lds/global/type, group1: dims).
// ---------------------------------------------------------------------------
__device__ __forceinline__ void tdm_load_1d(const void* gsrc, unsigned int lds_off,
                                            unsigned int n8) {
  unsigned long long ga = (unsigned long long)(uintptr_t)gsrc;
  uint32x4 g0;
  g0[0] = 1u;                                        // count=1, user descriptor
  g0[1] = lds_off;                                   // lds_addr (bytes)
  g0[2] = (unsigned int)ga;                          // global_addr[31:0]
  g0[3] = (unsigned int)((ga >> 32) & 0x01FFFFFFu)   // global_addr[56:32]
        | (2u << 30);                                // type=2 ("image")
  int32x8 g1;
  g1[0] = (int)(3u << 16);                           // data_size=3 (8 bytes)
  g1[1] = (int)(n8 << 16);                           // tensor_dim0[15:0]
  g1[2] = (int)((n8 >> 16) | (1u << 16));            // tensor_dim0[31:16] | tensor_dim1=1
  g1[3] = (int)(n8 << 16);                           // tile_dim0
  g1[4] = 0;                                         // tile_dim1=0, tile_dim2=0
  g1[5] = (int)n8;                                   // tensor_dim0_stride[31:0]
  g1[6] = 0;
  g1[7] = 0;
  int32x4 z4 = {0, 0, 0, 0};
#if __clang_major__ >= 23
  int32x8 z8 = {0, 0, 0, 0, 0, 0, 0, 0};
  __builtin_amdgcn_tensor_load_to_lds(g0, g1, z4, z4, z8, 0);
#else
  __builtin_amdgcn_tensor_load_to_lds(g0, g1, z4, z4, 0);
#endif
}

// ---------------------------------------------------------------------------
// Precompute: windowed DFT matrix in WMMA B-fragment layout (f16).
// B order used here: col N = lane&15, K = i + 16*(lane>>4) (contiguous k, same as A).
// Tiles 0..12  : Re coefficients  win[k]*cos(2*pi*f*k/400)
// Tiles 13..25 : Im coefficients  win[k]*sin(2*pi*f*k/400)  (sign dropped; squared)
// ---------------------------------------------------------------------------
__global__ void build_dft_frag(const float* __restrict__ win,
                               _Float16* __restrict__ frag) {
  int gid = blockIdx.x * blockDim.x + threadIdx.x;
  const int total = 2 * NT * KT * 32;
  if (gid >= total) return;
  int lane = gid & 31;
  int ks   = (gid >> 5) % KT;
  int tile = gid / (KT * 32);
  int hi = lane >> 4, ln = lane & 15;
  bool isSin = tile >= NT;
  int f = (isSin ? tile - NT : tile) * 16 + ln;
  v16h v;
#pragma unroll
  for (int i = 0; i < 16; ++i) {
    int k = ks * 32 + i + 16 * hi;              // B fragment K index
    float val = 0.f;
    if (k < N_FFT && f < N_FREQ) {
      int m = (f * k) % N_FFT;                  // exact integer phase reduction
      float ang = 6.28318530717958647692f * (float)m / (float)N_FFT;
      float tr = isSin ? sinf(ang) : cosf(ang);
      val = win[k] * tr;
    }
    v[i] = (_Float16)val;
  }
  *(v16h*)(frag + (size_t)gid * 16) = v;
}

// ---------------------------------------------------------------------------
// Precompute: mel filterbank (transposed: K=freq, N=mel) in B-fragment layout.
// ---------------------------------------------------------------------------
__global__ void build_mel_frag(const float* __restrict__ fb,
                               _Float16* __restrict__ frag) {
  int gid = blockIdx.x * blockDim.x + threadIdx.x;
  const int total = MNT * MKT * 32;
  if (gid >= total) return;
  int lane = gid & 31;
  int ks   = (gid >> 5) % MKT;
  int mt   = gid / (MKT * 32);
  int hi = lane >> 4, ln = lane & 15;
  int m = mt * 16 + ln;
  v16h v;
#pragma unroll
  for (int i = 0; i < 16; ++i) {
    int k = ks * 32 + i + 16 * hi;
    float val = (k < N_FREQ && m < N_MELS) ? fb[m * N_FREQ + k] : 0.f;
    v[i] = (_Float16)val;
  }
  *(v16h*)(frag + (size_t)gid * 16) = v;
}

// ---------------------------------------------------------------------------
// Fused: frames -> (windowed DFT via WMMA) -> |.|^2 -> (mel via WMMA) -> log.
// One WG = 1 batch x 64 frames.  4 waves, each owning a 16-frame M-tile.
// DFT B fragments double-buffered in LDS via Tensor Data Mover (TDM).
// ---------------------------------------------------------------------------
__global__ __launch_bounds__(128) void logmel_main(
    const float*    __restrict__ audio,
    const _Float16* __restrict__ dftfrag,
    const _Float16* __restrict__ melfrag,
    float*          __restrict__ out) {
  __shared__ float    sAudio[AUD_LDS];                      // 41.9 KB
  __shared__ _Float16 sPow[NWAVE][16][PFREQ];               // 28.7 KB
  __shared__ float    sOut[N_MELS][FPB];                    // 20.5 KB
  __shared__ __align__(16) _Float16 sB[2][2][TILE_HALFS];   // 53.2 KB (dbl buf)

  const int b    = blockIdx.x;
  const int fb   = blockIdx.y;                      // frame block (64 frames)
  const int tid  = threadIdx.x;
  const int wave = tid >> 5;
  const int lane = tid & 31;
  const int hi   = lane >> 4;
  const int ln   = lane & 15;

  // ---- prologue: TDM-prefetch cos/sin fragment blocks for freq tile 0 ------
  if (wave == 0) {
    tdm_load_1d(dftfrag,                  (unsigned int)(uintptr_t)&sB[0][0][0], TILE_HALFS / 4);
    tdm_load_1d(dftfrag + (size_t)NT * TILE_HALFS,
                                          (unsigned int)(uintptr_t)&sB[0][1][0], TILE_HALFS / 4);
  }

  // ---- stage audio into LDS with reflect padding applied -------------------
  const float* ab = audio + (size_t)b * T_LEN;
  const int base = fb * FPB * HOP;                  // start in padded domain
  for (int s = tid; s < AUD_LDS; s += 128) {
    int i = base + s - 200;                         // reflect: pad=200
    if (i < 0) i = -i;
    if (i >= T_LEN) i = 2 * T_LEN - 2 - i;
    sAudio[s] = ab[i];
  }
  __syncthreads();

  // ---- build all 13 A fragments (16 frames x 32 samples, f16) once ---------
  // A order used here: row M = lane&15, K = i + 16*(lane>>4) (contiguous k, same as B)
  const int frameLocal = wave * 16 + ln;
  v16h afrag[KT];
#pragma unroll
  for (int ks = 0; ks < KT; ++ks) {
    v16h a;
#pragma unroll
    for (int i = 0; i < 16; ++i) {
      int K = i + 16 * hi;
      int k = ks * 32 + K;
      float x = (k < N_FFT) ? sAudio[frameLocal * HOP + k] : 0.f;
      a[i] = (_Float16)x;
    }
    afrag[ks] = a;
  }

  // ---- DFT GEMM: per freq tile, accumulate Re and Im, then power -----------
  for (int nt = 0; nt < NT; ++nt) {
    const int cur = nt & 1;
    if (wave == 0) {
      if (nt + 1 < NT) {
        // prefetch next tile pair into the other buffer (it is idle now:
        // the trailing barrier of iteration nt-1 retired all its readers)
        tdm_load_1d(dftfrag + (size_t)(nt + 1) * TILE_HALFS,
                    (unsigned int)(uintptr_t)&sB[cur ^ 1][0][0], TILE_HALFS / 4);
        tdm_load_1d(dftfrag + (size_t)(nt + 1 + NT) * TILE_HALFS,
                    (unsigned int)(uintptr_t)&sB[cur ^ 1][1][0], TILE_HALFS / 4);
        __builtin_amdgcn_s_wait_tensorcnt(2);   // current pair complete
      } else {
        __builtin_amdgcn_s_wait_tensorcnt(0);
      }
    }
    __syncthreads();                            // publish sB[cur] to all waves

    v8f accRe = {};
    v8f accIm = {};
#pragma unroll
    for (int ks = 0; ks < KT; ++ks) {
      v16h bcf = *(const v16h*)&sB[cur][0][(ks * 32 + lane) * 16];
      v16h bsf = *(const v16h*)&sB[cur][1][(ks * 32 + lane) * 16];
      accRe = __builtin_amdgcn_wmma_f32_16x16x32_f16(
          false, afrag[ks], false, bcf, (short)0, accRe, false, false);
      accIm = __builtin_amdgcn_wmma_f32_16x16x32_f16(
          false, afrag[ks], false, bsf, (short)0, accIm, false, false);
    }
    // D-layout: reg r -> frame M = r + 8*hi; lane -> freq offset N = lane&15.
    // Re and Im live in the same lane/reg => per-lane power, no shuffles.
#pragma unroll
    for (int r = 0; r < 8; ++r) {
      float p = accRe[r] * accRe[r] + accIm[r] * accIm[r];
      sPow[wave][r + 8 * hi][nt * 16 + ln] = (_Float16)p;
    }
    __syncthreads();                            // all readers done before next TDM
  }
  // zero-pad freqs 208..223 (201..207 are already exact zeros via B padding)
  for (int z = lane; z < 16 * 16; z += 32)
    sPow[wave][z >> 4][208 + (z & 15)] = (_Float16)0.f;
  __syncthreads();

  // ---- mel GEMM: power(16x224) x melW^T(224x80) via WMMA -------------------
  for (int mt = 0; mt < MNT; ++mt) {
    v8f acc = {};
#pragma unroll
    for (int ks2 = 0; ks2 < MKT; ++ks2) {
      v16h a;
#pragma unroll
      for (int i = 0; i < 16; ++i) {
        int K = i + 16 * hi;
        a[i] = sPow[wave][ln][ks2 * 32 + K];
      }
      v16h bf = *(const v16h*)(melfrag + ((mt * MKT + ks2) * 32 + lane) * 16);
      acc = __builtin_amdgcn_wmma_f32_16x16x32_f16(
          false, a, false, bf, (short)0, acc, false, false);
    }
#pragma unroll
    for (int r = 0; r < 8; ++r)
      sOut[mt * 16 + ln][wave * 16 + r + 8 * hi] = acc[r];
  }
  __syncthreads();

  // ---- log + coalesced store: out[b][m][frame] -----------------------------
  const int fbase = fb * FPB;
  for (int idx = tid; idx < N_MELS * FPB; idx += 128) {
    int m  = idx / FPB;
    int fl = idx % FPB;
    int frame = fbase + fl;
    if (frame < N_FRAMES)
      out[((size_t)b * N_MELS + m) * N_FRAMES + frame] = logf(sOut[m][fl] + EPS);
  }
}

// ---------------------------------------------------------------------------
extern "C" void kernel_launch(void* const* d_in, const int* in_sizes, int n_in,
                              void* d_out, int out_size, void* d_ws, size_t ws_size,
                              hipStream_t stream) {
  const float* audio  = (const float*)d_in[0];   // [B, 160000] f32
  const float* window = (const float*)d_in[1];   // [400] f32
  const float* melfb  = (const float*)d_in[2];   // [80, 201] f32
  float* out = (float*)d_out;                    // [B, 80, 1001] f32

  const int B = in_sizes[0] / T_LEN;

  _Float16* dftfrag = (_Float16*)d_ws;                    // 173056 halves
  _Float16* melfrag = dftfrag + DFT_FRAG_HALFS;           //  17920 halves

  build_dft_frag<<<(2 * NT * KT * 32 + 255) / 256, 256, 0, stream>>>(window, dftfrag);
  build_mel_frag<<<(MNT * MKT * 32 + 255) / 256, 256, 0, stream>>>(melfb, melfrag);

  dim3 grid(B, (N_FRAMES + FPB - 1) / FPB);               // 128 x 16 WGs
  logmel_main<<<grid, 128, 0, stream>>>(audio, dftfrag, melfrag, out);
}
